// MultiheadAttention_83184926589115
// MI455X (gfx1250) — matrix operand order repaired and checked by reading, not compiled, not run on hardware
//
#include <hip/hip_runtime.h>
#include <hip/hip_bf16.h>
#include <cstdint>

// ---- problem constants ----
#define BB 2
#define SS 2048
#define FF 1024
#define HH 16
#define HD 64
#define MTOT (BB * SS)   // 4096

typedef unsigned int u32;
typedef __attribute__((ext_vector_type(16))) __bf16 v16bf;
typedef __attribute__((ext_vector_type(8)))  float  v8f;
typedef __attribute__((ext_vector_type(4)))  u32    u32x4;

union FragBF {           // 16x32 (A) or 32x16 (B) bf16 fragment, 8 VGPRs
    v16bf v;
    u32x4 q[2];
};

__device__ __forceinline__ v8f wmma_bf16(const FragBF& a, const FragBF& b, v8f c) {
    return __builtin_amdgcn_wmma_f32_16x16x32_bf16(
        /*neg_a=*/false, a.v, /*neg_b=*/false, b.v,
        /*c_mod=*/(short)0, c, /*reuse_a=*/false, /*reuse_b=*/false);
}

// ---------------------------------------------------------------------------
// Elementwise fp32 -> bf16 cast
// ---------------------------------------------------------------------------
__global__ void cast_bf16_k(const float* __restrict__ x, __bf16* __restrict__ y, int n) {
    int i = blockIdx.x * blockDim.x + threadIdx.x;
    if (i < n) y[i] = (__bf16)x[i];
}

// W [1024 x 1024] fp32 row-major  ->  Wt [1024 x 1024] bf16, Wt[n*K + k] = W[k*N + n]
__global__ void transpose_bf16_k(const float* __restrict__ W, __bf16* __restrict__ Wt) {
    int i = blockIdx.x * blockDim.x + threadIdx.x;   // over 1M
    int k = i >> 10, n = i & 1023;
    Wt[(size_t)n * 1024 + k] = (__bf16)W[i];
}

// RoPE multiplier table emb[S][64]: concat[sin,cos], symmetric-averaged for s>0
__global__ void emb_k(float* __restrict__ emb) {
    int i = blockIdx.x * blockDim.x + threadIdx.x;   // over S*64
    int s = i >> 6, d = i & 63, j = d & 31;
    float inv = powf(10000.0f, -(float)j * (1.0f / 32.0f));
    float a1 = (float)s * inv;
    float v1 = (d < 32) ? sinf(a1) : cosf(a1);
    float out = v1;
    if (s > 0) {
        float a2 = (float)(SS - s) * inv;
        float v2 = (d < 32) ? sinf(a2) : cosf(a2);
        out = 0.5f * (v1 + v2);
    }
    emb[i] = out;
}

// ---------------------------------------------------------------------------
// bf16 WMMA GEMM:  C[M=4096, N=1024] = A[4096 x 1024] * Bt^T   (Bt is [N x K])
// MODE 0: Q proj  -> rope multiply, store bf16 [B,H,S,HD]
// MODE 1: K proj  -> rope multiply, store bf16 [B,H,S,HD]
// MODE 2: V proj  -> store bf16 transposed [B,H,HD,S]
// MODE 3: out proj -> store fp32 [M, N]
// Block: 128 threads = 4 waves in 2x2, block tile 128x128, wave tile 64x64.
// Per K-step (32): 8 fragment loads feed 16 WMMAs (A frags reused 4x each).
// ---------------------------------------------------------------------------
template <int MODE>
__global__ __launch_bounds__(128)
void gemm_bf16_k(const __bf16* __restrict__ A, const __bf16* __restrict__ Bt,
                 void* __restrict__ Dst, const float* __restrict__ emb) {
    const int K = FF;
    const int lane = threadIdx.x & 31;
    const int w    = threadIdx.x >> 5;
    const int half = lane >> 4;
    const int l15  = lane & 15;
    const int wm = blockIdx.x * 128 + (w >> 1) * 64;
    const int wn = blockIdx.y * 128 + (w & 1) * 64;

    v8f c[4][4] = {};
    const __bf16* arow[4];
    const __bf16* bcol[4];
#pragma unroll
    for (int i = 0; i < 4; ++i) {
        arow[i] = A  + (size_t)(wm + i * 16 + l15) * K;
        bcol[i] = Bt + (size_t)(wn + i * 16 + l15) * K;
    }

    for (int k0 = 0; k0 < K; k0 += 32) {
        __builtin_prefetch(arow[0] + k0 + 128, 0, 1);
        __builtin_prefetch(bcol[0] + k0 + 128, 0, 1);
        FragBF a[4], b[4];
#pragma unroll
        for (int i = 0; i < 4; ++i) {
            const __bf16* p = arow[i] + k0 + half * 16;  // A: k = kg*16 + e, contiguous
            a[i].q[0] = *(const u32x4*)(p);
            a[i].q[1] = *(const u32x4*)(p + 8);
        }
#pragma unroll
        for (int j = 0; j < 4; ++j) {
            const __bf16* p = bcol[j] + k0 + half * 16;  // B: k = kg*16 + e, contiguous
            b[j].q[0] = *(const u32x4*)(p);
            b[j].q[1] = *(const u32x4*)(p + 8);
        }
#pragma unroll
        for (int i = 0; i < 4; ++i)
#pragma unroll
            for (int j = 0; j < 4; ++j)
                c[i][j] = wmma_bf16(a[i], b[j], c[i][j]);
    }

#pragma unroll
    for (int i = 0; i < 4; ++i)
#pragma unroll
        for (int j = 0; j < 4; ++j)
#pragma unroll
            for (int r = 0; r < 8; ++r) {
                int m = wm + i * 16 + r + half * 8;      // C/D layout: VGPR r -> M=r(+8)
                int n = wn + j * 16 + l15;               // lane -> N
                float v = c[i][j][r];
                if (MODE == 3) {
                    ((float*)Dst)[(size_t)m * FF + n] = v;
                } else {
                    int b_ = m >> 11, s = m & (SS - 1);
                    int h  = n >> 6,  d = n & (HD - 1);
                    if (MODE == 0 || MODE == 1) v *= emb[s * HD + d];
                    __bf16* D = (__bf16*)Dst;
                    size_t bh = (size_t)(b_ * HH + h);
                    if (MODE == 2) D[(bh * HD + d) * SS + s] = (__bf16)v;     // V^T
                    else           D[(bh * SS + s) * HD + d] = (__bf16)v;     // Q / K
                }
            }
}

// ---------------------------------------------------------------------------
// Causal flash attention: one wave per (b, h, 16-row q block) = 4096 waves.
// Q,K: bf16 [B,H,S,HD]; Vt: bf16 [B,H,HD,S]; Ob: bf16 [B,S,F] (attn output).
// ---------------------------------------------------------------------------
__global__ __launch_bounds__(32)
void attn_k(const __bf16* __restrict__ Q, const __bf16* __restrict__ Kh,
            const __bf16* __restrict__ Vt, __bf16* __restrict__ Ob) {
    __shared__ __align__(16) __bf16 plds[16 * 32];   // P tile relayout buffer

    const int lane = threadIdx.x & 31;
    const int half = lane >> 4;
    const int l15  = lane & 15;
    const int qb = blockIdx.x & 127;            // S/16 blocks
    const int h  = (blockIdx.x >> 7) & 15;
    const int b  = blockIdx.x >> 11;
    const size_t bh = (size_t)(b * HH + h);

    // Q rows as two A fragments (d = 0..31, 32..63), resident for whole loop
    FragBF aq[2];
    {
        const __bf16* qp = Q + (bh * SS + qb * 16 + l15) * HD;
        aq[0].q[0] = *(const u32x4*)(qp + half * 16);
        aq[0].q[1] = *(const u32x4*)(qp + half * 16 + 8);
        aq[1].q[0] = *(const u32x4*)(qp + 32 + half * 16);
        aq[1].q[1] = *(const u32x4*)(qp + 32 + half * 16 + 8);
    }

    v8f o[4] = {};                // 16 x 64 output accumulator (4 N-tiles)
    float mrow[8], lrow[8];
#pragma unroll
    for (int r = 0; r < 8; ++r) { mrow[r] = -3.0e38f; lrow[r] = 0.0f; }

    const int jmax = (qb * 16 + 15) >> 5;       // last 32-col chunk touching diagonal
    for (int j = 0; j <= jmax; ++j) {
        // ---- scores: two 16x16 tiles over 32 key columns ----
        v8f s[2];
#pragma unroll
        for (int t = 0; t < 2; ++t) {
            const __bf16* kp = Kh + (bh * SS + j * 32 + t * 16 + l15) * HD;
            FragBF bk0, bk1;                     // d = 0..31 and 32..63 slices
            bk0.q[0] = *(const u32x4*)(kp + half * 16);
            bk0.q[1] = *(const u32x4*)(kp + half * 16 + 8);
            bk1.q[0] = *(const u32x4*)(kp + 32 + half * 16);
            bk1.q[1] = *(const u32x4*)(kp + 32 + half * 16 + 8);
            v8f z = {};
            s[t] = wmma_bf16(aq[0], bk0, z);
            s[t] = wmma_bf16(aq[1], bk1, s[t]);
        }
        // ---- scale + causal mask ----
#pragma unroll
        for (int t = 0; t < 2; ++t)
#pragma unroll
            for (int r = 0; r < 8; ++r) {
                float v = s[t][r] * 0.125f;      // 1/sqrt(64)
                int kcol = j * 32 + t * 16 + l15;
                int qrow = qb * 16 + r + half * 8;
                if (kcol > qrow) v = -3.0e38f;
                s[t][r] = v;
            }
        // ---- online softmax, per row (rows live in 16-lane half-groups) ----
#pragma unroll
        for (int r = 0; r < 8; ++r) {
            float rm = fmaxf(s[0][r], s[1][r]);
            rm = fmaxf(rm, __shfl_xor(rm, 1, 32));
            rm = fmaxf(rm, __shfl_xor(rm, 2, 32));
            rm = fmaxf(rm, __shfl_xor(rm, 4, 32));
            rm = fmaxf(rm, __shfl_xor(rm, 8, 32));
            float nm    = fmaxf(mrow[r], rm);
            float alpha = __expf(mrow[r] - nm);
            float p0    = __expf(s[0][r] - nm);
            float p1    = __expf(s[1][r] - nm);
            float rs = p0 + p1;
            rs += __shfl_xor(rs, 1, 32);
            rs += __shfl_xor(rs, 2, 32);
            rs += __shfl_xor(rs, 4, 32);
            rs += __shfl_xor(rs, 8, 32);
            lrow[r] = lrow[r] * alpha + rs;
            mrow[r] = nm;
#pragma unroll
            for (int t4 = 0; t4 < 4; ++t4) o[t4][r] *= alpha;
            int m = r + half * 8;
            plds[m * 32 + l15]      = (__bf16)p0;
            plds[m * 32 + 16 + l15] = (__bf16)p1;
        }
        __asm__ volatile("" ::: "memory");       // keep DS writes before DS reads
        // ---- relayout P (C-frag) -> A-frag via LDS (wave-private, in-order DS) ----
        FragBF pa;
        {
            const __bf16* pp = plds + l15 * 32 + half * 16;
            pa.q[0] = *(const u32x4*)(pp);
            pa.q[1] = *(const u32x4*)(pp + 8);
        }
        __asm__ volatile("" ::: "memory");
        // ---- O += P * V  (V pre-transposed -> contiguous B fragments) ----
#pragma unroll
        for (int t4 = 0; t4 < 4; ++t4) {
            const __bf16* vp = Vt + (bh * HD + t4 * 16 + l15) * SS + j * 32 + half * 16;
            FragBF bv;
            bv.q[0] = *(const u32x4*)(vp);
            bv.q[1] = *(const u32x4*)(vp + 8);
            o[t4] = wmma_bf16(pa, bv, o[t4]);
        }
    }

    // ---- normalize and store attn output as bf16 [B, S, F] ----
#pragma unroll
    for (int r = 0; r < 8; ++r) {
        float inv_l = 1.0f / lrow[r];
        int s_ = qb * 16 + r + half * 8;
#pragma unroll
        for (int t4 = 0; t4 < 4; ++t4) {
            float v = o[t4][r] * inv_l;
            int d = t4 * 16 + l15;
            Ob[((size_t)(b * SS + s_)) * FF + h * HD + d] = (__bf16)v;
        }
    }
}

// ---------------------------------------------------------------------------
extern "C" void kernel_launch(void* const* d_in, const int* in_sizes, int n_in,
                              void* d_out, int out_size, void* d_ws, size_t ws_size,
                              hipStream_t stream) {
    const float* kv = (const float*)d_in[0];
    const float* q  = (const float*)d_in[1];
    const float* Wq = (const float*)d_in[2];
    const float* Wk = (const float*)d_in[3];
    const float* Wv = (const float*)d_in[4];
    const float* Wo = (const float*)d_in[5];
    // d_in[6] = mask: causal tril, applied analytically in attn_k.

    char* ws = (char*)d_ws;
    size_t off = 0;
    auto alloc = [&](size_t bytes) -> void* {
        void* p = ws + off;
        off += (bytes + 255) & ~(size_t)255;
        return p;
    };
    const size_t nAct = (size_t)MTOT * FF;   // 4M elements
    const size_t nW   = (size_t)FF * FF;     // 1M elements
    __bf16* qb    = (__bf16*)alloc(nAct * 2);
    __bf16* kvb   = (__bf16*)alloc(nAct * 2);
    __bf16* Wqt   = (__bf16*)alloc(nW * 2);
    __bf16* Wkt   = (__bf16*)alloc(nW * 2);
    __bf16* Wvt   = (__bf16*)alloc(nW * 2);
    __bf16* Wot   = (__bf16*)alloc(nW * 2);
    __bf16* Qh    = (__bf16*)alloc(nAct * 2);   // [B,H,S,HD]
    __bf16* Kh    = (__bf16*)alloc(nAct * 2);   // [B,H,S,HD]
    __bf16* Vt    = (__bf16*)alloc(nAct * 2);   // [B,H,HD,S]
    __bf16* attnb = (__bf16*)alloc(nAct * 2);   // [B,S,F]
    float*  embf  = (float*)alloc((size_t)SS * HD * 4);

    cast_bf16_k<<<(int)((nAct + 255) / 256), 256, 0, stream>>>(q,  qb,  (int)nAct);
    cast_bf16_k<<<(int)((nAct + 255) / 256), 256, 0, stream>>>(kv, kvb, (int)nAct);
    transpose_bf16_k<<<(int)(nW / 256), 256, 0, stream>>>(Wq, Wqt);
    transpose_bf16_k<<<(int)(nW / 256), 256, 0, stream>>>(Wk, Wkt);
    transpose_bf16_k<<<(int)(nW / 256), 256, 0, stream>>>(Wv, Wvt);
    transpose_bf16_k<<<(int)(nW / 256), 256, 0, stream>>>(Wo, Wot);
    emb_k<<<(SS * HD) / 256, 256, 0, stream>>>(embf);

    dim3 gg(MTOT / 128, FF / 128), gb(128);
    gemm_bf16_k<0><<<gg, gb, 0, stream>>>(qb,  Wqt, (void*)Qh,  embf);
    gemm_bf16_k<1><<<gg, gb, 0, stream>>>(kvb, Wkt, (void*)Kh,  embf);
    gemm_bf16_k<2><<<gg, gb, 0, stream>>>(kvb, Wvt, (void*)Vt,  embf);

    attn_k<<<BB * HH * (SS / 16), 32, 0, stream>>>(Qh, Kh, Vt, attnb);

    gemm_bf16_k<3><<<gg, gb, 0, stream>>>(attnb, Wot, d_out, embf);
}
